// Compressor_20315195310384
// MI455X (gfx1250) — compile-verified, matrix operand order checked by reading, not run on hardware
//
#include <hip/hip_runtime.h>
#include <hip/hip_bf16.h>

typedef __attribute__((ext_vector_type(16))) __bf16 v16bf;
typedef __attribute__((ext_vector_type(8)))  __bf16 v8bf;
typedef __attribute__((ext_vector_type(4)))  __bf16 v4bf;
typedef __attribute__((ext_vector_type(8)))  float  v8f;

#define TOKENS 16384   // b*s = 4*4096
#define DIMK   4096
#define NOUT   2048    // [0,1024)=kv  [1024,2048)=gate
#define BM 128
#define BN 128
#define BK 32
#define APAD 40        // row stride in bf16 elems: 80B, 16B aligned, bank-rotating

#define HEAD 512
#define NOPE 448

// ---------------------------------------------------------------------------
// GEMM: C[16384,2048] = X[16384,4096] x [W_kv;W_gate]^T   (bf16 WMMA, f32 acc)
// Workgroup tile 128x128, wave tile 64x32 -> 8 WMMA per 12 ds_load_b128.
// ---------------------------------------------------------------------------
__global__ __launch_bounds__(256) void gemm_bf16_wmma(
    const float* __restrict__ X, const float* __restrict__ Wkv,
    const float* __restrict__ Wgate, float* __restrict__ C)
{
    __shared__ __bf16 As[BM][APAD];
    __shared__ __bf16 Bs[BN][APAD];

    const int tid  = threadIdx.x;
    const int nblk = blockIdx.x & 15;          // 16 n-blocks innermost: X panel reuse
    const int mblk = blockIdx.x >> 4;          // 128 m-blocks
    const int m0 = mblk * BM;
    const int n0 = nblk * BN;

    const int wave  = tid >> 5;
    const int lane  = tid & 31;
    const int waveM = wave & 1;                // 0..1  (64-row patch)
    const int waveN = wave >> 1;               // 0..3  (32-col patch)
    const int half  = lane >> 4;               // 0/1
    const int l16   = lane & 15;

    v8f acc[4][2] = {};

    for (int kt = 0; kt < DIMK; kt += BK) {
        // ---- stage A tile (128x32 f32 -> bf16) ----
        #pragma unroll
        for (int r = 0; r < 4; ++r) {
            int idx = tid + r * 256;                       // 1024 float4 slots
            int row = idx >> 3;
            int c4  = (idx & 7) << 2;
            const float4 v = *(const float4*)(X + (size_t)(m0 + row) * DIMK + kt + c4);
            v4bf bv = { (__bf16)v.x, (__bf16)v.y, (__bf16)v.z, (__bf16)v.w };
            *(v4bf*)&As[row][c4] = bv;
        }
        // ---- stage B tile (128x32 f32 -> bf16), rows = output cols ----
        // n0 is a multiple of 128, so the whole tile is either kv or gate (uniform).
        const float* Wbase = (n0 < 1024) ? (Wkv + (size_t)n0 * DIMK)
                                         : (Wgate + (size_t)(n0 - 1024) * DIMK);
        #pragma unroll
        for (int r = 0; r < 4; ++r) {
            int idx = tid + r * 256;                       // 1024 float4 slots
            int row = idx >> 3;
            int c4  = (idx & 7) << 2;
            const float4 v = *(const float4*)(Wbase + (size_t)row * DIMK + kt + c4);
            v4bf bv = { (__bf16)v.x, (__bf16)v.y, (__bf16)v.z, (__bf16)v.w };
            *(v4bf*)&Bs[row][c4] = bv;
        }
        if (kt + BK < DIMK)   // hint next K tile of X toward the caches
            __builtin_prefetch(X + (size_t)(m0 + (tid >> 1)) * DIMK + kt + BK, 0, 1);
        __syncthreads();

        // ---- fragments per ISA layout: lanes0-15 K{0..7,16..23}, lanes16-31 K{8..15,24..31}
        v16bf a[4], b[2];
        #pragma unroll
        for (int mi = 0; mi < 4; ++mi) {
            int arow = waveM * 64 + mi * 16 + l16;
            v8bf lo = *(const v8bf*)&As[arow][half * 8];
            v8bf hi = *(const v8bf*)&As[arow][16 + half * 8];
            #pragma unroll
            for (int i = 0; i < 8; ++i) { a[mi][i] = lo[i]; a[mi][i + 8] = hi[i]; }
        }
        #pragma unroll
        for (int ni = 0; ni < 2; ++ni) {
            int brow = waveN * 32 + ni * 16 + l16;
            v8bf lo = *(const v8bf*)&Bs[brow][half * 8];
            v8bf hi = *(const v8bf*)&Bs[brow][16 + half * 8];
            #pragma unroll
            for (int i = 0; i < 8; ++i) { b[ni][i] = lo[i]; b[ni][i + 8] = hi[i]; }
        }
        #pragma unroll
        for (int mi = 0; mi < 4; ++mi)
            #pragma unroll
            for (int ni = 0; ni < 2; ++ni)
                acc[mi][ni] = __builtin_amdgcn_wmma_f32_16x16x32_bf16(
                    false, a[mi], false, b[ni], (short)0, acc[mi][ni], false, false);
        __syncthreads();
    }

    // ---- write C: VGPR v -> M = half*8 + v, N = l16 ----
    #pragma unroll
    for (int mi = 0; mi < 4; ++mi)
        #pragma unroll
        for (int ni = 0; ni < 2; ++ni) {
            int mbase = m0 + waveM * 64 + mi * 16 + half * 8;
            int nn    = n0 + waveN * 32 + ni * 16 + l16;
            #pragma unroll
            for (int v = 0; v < 8; ++v)
                C[(size_t)(mbase + v) * NOUT + nn] = acc[mi][ni][v];
        }
}

// ---------------------------------------------------------------------------
// Epilogue: overlap softmax-pool, RMSNorm, RoPE, FWHT (Hadamard/sqrt(512))
// One workgroup (256 thr) per (batch, block); each thread owns 2 columns.
// ---------------------------------------------------------------------------
__global__ __launch_bounds__(256) void epilogue_kernel(
    const float* __restrict__ C, const float* __restrict__ ape,
    const float* __restrict__ norm_w, const float* __restrict__ freqs,
    float* __restrict__ out)
{
    __shared__ float sp[HEAD];
    __shared__ float red[256];

    const int tid = threadIdx.x;
    const int bb  = blockIdx.x >> 10;           // batch 0..3
    const int blk = blockIdx.x & 1023;          // block 0..1023
    const size_t tok_cur = (size_t)bb * 4096 + (size_t)blk * 4;
    const bool has_prev = (blk != 0);

    float p[2];
    #pragma unroll
    for (int cc = 0; cc < 2; ++cc) {
        const int c = tid * 2 + cc;
        float kvv[8], sc[8];
        #pragma unroll
        for (int j = 0; j < 4; ++j) {          // rows 0..3: prev block, cols [0,512)
            if (has_prev) {
                size_t t = (tok_cur - 4 + j) * NOUT;
                kvv[j] = C[t + c] + ape[j * 1024 + c];
                sc[j]  = C[t + 1024 + c];
            } else { kvv[j] = 0.0f; sc[j] = -1e30f; }
        }
        #pragma unroll
        for (int j = 0; j < 4; ++j) {          // rows 4..7: cur block, cols [512,1024)
            size_t t = (tok_cur + j) * NOUT;
            kvv[4 + j] = C[t + 512 + c] + ape[j * 1024 + 512 + c];
            sc[4 + j]  = C[t + 1024 + 512 + c];
        }
        float m = sc[0];
        #pragma unroll
        for (int j = 1; j < 8; ++j) m = fmaxf(m, sc[j]);
        float den = 0.0f, num = 0.0f;
        #pragma unroll
        for (int j = 0; j < 8; ++j) {
            float e = __expf(sc[j] - m);
            den += e; num += e * kvv[j];
        }
        p[cc] = num / den;
    }

    // RMS: sum of squares over 512 (LDS tree)
    red[tid] = p[0] * p[0] + p[1] * p[1];
    __syncthreads();
    for (int s = 128; s > 0; s >>= 1) {
        if (tid < s) red[tid] += red[tid + s];
        __syncthreads();
    }
    const float r = rsqrtf(red[0] / (float)HEAD + 1e-6f);
    #pragma unroll
    for (int cc = 0; cc < 2; ++cc) {
        int c = tid * 2 + cc;
        p[cc] = norm_w[c] * (p[cc] * r);
    }

    // RoPE: threads 224..255 own exactly one (even,odd) pair of the last 64 dims
    if (tid >= NOPE / 2) {
        int k = tid - NOPE / 2;                 // pair 0..31
        float fr = freqs[(size_t)blk * 64 + k * 2 + 0];
        float fi = freqs[(size_t)blk * 64 + k * 2 + 1];
        float xr = p[0], xi = p[1];
        p[0] = xr * fr - xi * fi;
        p[1] = xr * fi + xi * fr;
    }
    sp[tid * 2]     = p[0];
    sp[tid * 2 + 1] = p[1];

    // FWHT (Sylvester Hadamard is symmetric: v@H == H@v == FWHT(v)/sqrt(512))
    for (int h = 1; h < HEAD; h <<= 1) {
        __syncthreads();
        int i = ((tid / h) * (h << 1)) + (tid % h);
        float va = sp[i], vb = sp[i + h];
        sp[i] = va + vb; sp[i + h] = va - vb;
    }
    __syncthreads();

    const float scale = 0.044194173824159216f;  // 1/sqrt(512)
    size_t obase = (size_t)blockIdx.x * HEAD;
    out[obase + tid * 2]     = sp[tid * 2]     * scale;
    out[obase + tid * 2 + 1] = sp[tid * 2 + 1] * scale;
}

// ---------------------------------------------------------------------------
extern "C" void kernel_launch(void* const* d_in, const int* in_sizes, int n_in,
                              void* d_out, int out_size, void* d_ws, size_t ws_size,
                              hipStream_t stream)
{
    (void)in_sizes; (void)n_in; (void)out_size; (void)ws_size;
    const float* x     = (const float*)d_in[0];
    const float* Wkv   = (const float*)d_in[1];
    const float* Wgate = (const float*)d_in[2];
    const float* ape   = (const float*)d_in[3];
    const float* normw = (const float*)d_in[4];
    const float* freqs = (const float*)d_in[5];
    float* C   = (float*)d_ws;                  // 16384*2048 f32 = 128 MB scratch
    float* out = (float*)d_out;

    gemm_bf16_wmma<<<dim3((TOKENS / BM) * (NOUT / BN)), 256, 0, stream>>>(x, Wkv, Wgate, C);
    epilogue_kernel<<<dim3(4 * 1024), 256, 0, stream>>>(C, ape, normw, freqs, out);
}
